// WindowCrossAttentionFusion_75007308857631
// MI455X (gfx1250) — matrix operand order repaired and checked by reading, not compiled, not run on hardware
//
#include <hip/hip_runtime.h>

// CDNA5 / gfx1250 fused window cross-attention.
// One workgroup (256 threads = 8 wave32) per 8x8 window; all activations in
// LDS as bf16; all GEMMs via v_wmma_f32_16x16x32_bf16; weights pre-converted
// to bf16 in workspace (stay hot in the 192MB L2 across 2048 blocks).
// Weight GEMMs: each wave owns 2 M-strips x 4 N-tiles; the global B-fragment
// stream is ping-pong double-buffered so the 8-WMMA chain of step k overlaps
// the loads of step k+1 (no s_wait_loadcnt 0x0 in steady state).

typedef __attribute__((ext_vector_type(16))) __bf16 v16bf;
typedef __attribute__((ext_vector_type(8)))  float  v8f;

#define DIM      256
#define HEADS    4
#define HD       64
#define NTOK     64          // 8x8 window tokens
#define IMG      128
#define NWIN     2048        // 8 images * 256 windows

// LDS row strides (elements) — padded to stay 16B-aligned and spread banks
#define AST 264              // for 64x256 tiles (a, b, q, k, o)
#define VST 72               // for 256x64 v-transposed
#define PST 72               // for per-head 64x64 probabilities

union FragU { v16bf v; uint4 q[2]; };

static __device__ __forceinline__ v8f wmma_bf16(v16bf a, v16bf b, v8f c) {
  return __builtin_amdgcn_wmma_f32_16x16x32_bf16(
      /*neg_a=*/false, a, /*neg_b=*/false, b,
      /*c_mod=*/(short)0, c, /*reuse_a=*/false, /*reuse_b=*/false);
}

// A fragment: 16x32 bf16, rows row0..row0+15, K = k0..k0+31.
// Lane layout per ISA 7.12.2: lanes<16 hold K {k0..k0+7, k0+16..k0+23},
// lanes>=16 hold K {k0+8..k0+15, k0+24..k0+31}.
static __device__ __forceinline__ v16bf load_a(const __bf16* base, int row0, int stride, int k0) {
  int lane = threadIdx.x & 31;
  int l = lane & 15, hi = lane >> 4;
  const __bf16* p = base + (size_t)(row0 + l) * stride + k0 + hi * 8;
  FragU u;
  u.q[0] = *(const uint4*)(p);        // 8 bf16: K offsets +0..7
  u.q[1] = *(const uint4*)(p + 16);   // 8 bf16: K offsets +16..23
  return u.v;
}

// B fragment: 32x16 where B[k][n] = base[n][k] (base row-major, rows = output
// columns). Lanes<16 hold K {k0..k0+7, k0+16..k0+23}, lanes>=16 hold K
// {k0+8..k0+15, k0+24..k0+31}; column is lane&15.
static __device__ __forceinline__ v16bf load_b(const __bf16* base, int col0, int stride, int k0) {
  int lane = threadIdx.x & 31;
  int l = lane & 15, hi = lane >> 4;
  const __bf16* p = base + (size_t)(col0 + l) * stride + k0 + hi * 8;
  FragU u;
  u.q[0] = *(const uint4*)(p);
  u.q[1] = *(const uint4*)(p + 16);
  return u.v;
}

// 64x256 = (64x256 bf16 in LDS) x (W^T, W row-major 256x256 bf16) + bias.
// Wave w: M-strips m0=(w>>2)*2, m0+1 ; N-tiles (w&3)*4 .. +3.
// B stream double-buffered (bfA/bfB) across fully unrolled k-steps.
static __device__ __forceinline__ void gemm_proj(const __bf16* __restrict__ src,
                                                 const __bf16* __restrict__ W,
                                                 const float*  __restrict__ bias,
                                                 __bf16* __restrict__ dst,
                                                 int transpose_dst) {
  const int wave = threadIdx.x >> 5;
  const int lane = threadIdx.x & 31;
  const int l = lane & 15, hi = lane >> 4;
  const int m0 = (wave >> 2) * 2;        // M-strips m0, m0+1
  const int n0 = (wave & 3) * 4;         // N-tiles n0 .. n0+3
  const v8f vzero = {0.f,0.f,0.f,0.f,0.f,0.f,0.f,0.f};
  v8f acc0[4], acc1[4];
#pragma unroll
  for (int j = 0; j < 4; ++j) { acc0[j] = vzero; acc1[j] = vzero; }

  v16bf bfA[4], bfB[4];
#pragma unroll
  for (int j = 0; j < 4; ++j) bfA[j] = load_b(W, (n0 + j) * 16, DIM, 0);

#pragma unroll
  for (int kk = 0; kk < 8; kk += 2) {
    // even step: consume bfA, load bfB for kk+1
    v16bf af0 = load_a(src, (m0 + 0) * 16, AST, kk * 32);
    v16bf af1 = load_a(src, (m0 + 1) * 16, AST, kk * 32);
#pragma unroll
    for (int j = 0; j < 4; ++j) bfB[j] = load_b(W, (n0 + j) * 16, DIM, (kk + 1) * 32);
    if (kk + 2 < 8)  // prefetch two steps ahead (global_prefetch_b8)
      __builtin_prefetch(W + (size_t)(n0 * 16 + l) * DIM + (kk + 2) * 32, 0, 3);
#pragma unroll
    for (int j = 0; j < 4; ++j) {
      acc0[j] = wmma_bf16(af0, bfA[j], acc0[j]);
      acc1[j] = wmma_bf16(af1, bfA[j], acc1[j]);
    }
    // odd step: consume bfB, load bfA for kk+2
    af0 = load_a(src, (m0 + 0) * 16, AST, (kk + 1) * 32);
    af1 = load_a(src, (m0 + 1) * 16, AST, (kk + 1) * 32);
    if (kk + 2 < 8) {
#pragma unroll
      for (int j = 0; j < 4; ++j) bfA[j] = load_b(W, (n0 + j) * 16, DIM, (kk + 2) * 32);
    }
#pragma unroll
    for (int j = 0; j < 4; ++j) {
      acc0[j] = wmma_bf16(af0, bfB[j], acc0[j]);
      acc1[j] = wmma_bf16(af1, bfB[j], acc1[j]);
    }
  }

#pragma unroll
  for (int j = 0; j < 4; ++j) {
    int c = (n0 + j) * 16 + l;
    float bz = bias[c];
    if (!transpose_dst) {
#pragma unroll
      for (int r = 0; r < 8; ++r) {
        int t0 = (m0 + 0) * 16 + r + 8 * hi;
        int t1 = (m0 + 1) * 16 + r + 8 * hi;
        dst[t0 * AST + c] = (__bf16)(acc0[j][r] + bz);
        dst[t1 * AST + c] = (__bf16)(acc1[j][r] + bz);
      }
    } else {
#pragma unroll
      for (int r = 0; r < 8; ++r) {
        int t0 = (m0 + 0) * 16 + r + 8 * hi;
        int t1 = (m0 + 1) * 16 + r + 8 * hi;
        dst[c * VST + t0] = (__bf16)(acc0[j][r] + bz);   // v stored transposed
        dst[c * VST + t1] = (__bf16)(acc1[j][r] + bz);
      }
    }
  }
}

extern "C" __global__ void __launch_bounds__(256)
win_attn_kernel(const float* __restrict__ fa, const float* __restrict__ fb,
                const __bf16* __restrict__ Wq, const float* __restrict__ bq,
                const __bf16* __restrict__ Wk, const float* __restrict__ bk,
                const __bf16* __restrict__ Wv, const float* __restrict__ bv,
                const __bf16* __restrict__ Wo, const float* __restrict__ bo,
                float* __restrict__ out, float* __restrict__ attn_out) {
  extern __shared__ __bf16 smem[];
  __bf16* a_s = smem;                       // 64 x AST
  __bf16* b_s = a_s + NTOK * AST;           // 64 x AST
  __bf16* q_s = b_s + NTOK * AST;           // 64 x AST
  __bf16* k_s = q_s + NTOK * AST;           // 64 x AST
  __bf16* vT  = k_s + NTOK * AST;           // 256 x VST (v transposed)
  __bf16* p_s = vT  + DIM  * VST;           // HEADS x 64 x PST
  __bf16* o_s = p_s + HEADS * NTOK * PST;   // 64 x AST

  const int win  = blockIdx.x;
  const int bimg = win >> 8;
  const int rem  = win & 255;
  const int y0 = (rem >> 4) * 8;
  const int x0 = (rem & 15) * 8;
  const size_t img_base = (size_t)bimg * DIM * IMG * IMG;

  // ---- Phase A: gather both feature windows, convert f32 -> bf16 in LDS ----
  for (int i = threadIdx.x; i < NTOK * DIM; i += 256) {
    int t = i & 63;                 // token fastest -> x-contiguous global reads
    int c = i >> 6;
    int y = y0 + (t >> 3), x = x0 + (t & 7);
    size_t gi = img_base + ((size_t)c * IMG + y) * IMG + x;
    a_s[t * AST + c] = (__bf16)fa[gi];
    b_s[t * AST + c] = (__bf16)fb[gi];
  }
  __syncthreads();

  // ---- Phase B: Q/K/V projections (bf16 WMMA, f32 accum) ----
  gemm_proj(a_s, Wq, bq, q_s, 0);
  gemm_proj(b_s, Wk, bk, k_s, 0);
  gemm_proj(b_s, Wv, bv, vT, 1);
  __syncthreads();

  const int wave = threadIdx.x >> 5;
  const int lane = threadIdx.x & 31;
  const int l = lane & 15, hi = lane >> 4;
  const v8f vzero = {0.f,0.f,0.f,0.f,0.f,0.f,0.f,0.f};

  // ---- Phase C: S = (Q K^T) * scale, row softmax, emit attn + bf16 probs ----
  // 16 (head, m-tile) strips; each wave owns 2 whole strips -> wave-local softmax.
  for (int j = 0; j < 2; ++j) {
    int pid = wave * 2 + j;
    int h  = pid >> 2;
    int mt = pid & 3;
    v8f acc[4];
#pragma unroll
    for (int nt = 0; nt < 4; ++nt) acc[nt] = vzero;
#pragma unroll
    for (int kk = 0; kk < 2; ++kk) {
      v16bf qa = load_a(q_s, mt * 16, AST, h * HD + kk * 32);   // hoisted
      v16bf kb[4];
#pragma unroll
      for (int nt = 0; nt < 4; ++nt)
        kb[nt] = load_b(k_s, nt * 16, AST, h * HD + kk * 32);   // B[k][n] = K[n][k]
#pragma unroll
      for (int nt = 0; nt < 4; ++nt) acc[nt] = wmma_bf16(qa, kb[nt], acc[nt]);
    }
    const float scale = 0.125f;   // HEAD_DIM^-0.5
#pragma unroll
    for (int r = 0; r < 8; ++r) {
#pragma unroll
      for (int nt = 0; nt < 4; ++nt) acc[nt][r] *= scale;
      float m = fmaxf(fmaxf(acc[0][r], acc[1][r]), fmaxf(acc[2][r], acc[3][r]));
      for (int d = 1; d < 16; d <<= 1) m = fmaxf(m, __shfl_xor(m, d, 16));
      float s = 0.f;
#pragma unroll
      for (int nt = 0; nt < 4; ++nt) { acc[nt][r] = __expf(acc[nt][r] - m); s += acc[nt][r]; }
      for (int d = 1; d < 16; d <<= 1) s += __shfl_xor(s, d, 16);
      float inv = 1.0f / s;
      int mrow = mt * 16 + r + 8 * hi;
#pragma unroll
      for (int nt = 0; nt < 4; ++nt) {
        float pv = acc[nt][r] * inv;
        int ncol = nt * 16 + l;
        attn_out[(((size_t)win * HEADS + h) * NTOK + mrow) * NTOK + ncol] = pv;
        p_s[(h * NTOK + mrow) * PST + ncol] = (__bf16)pv;
      }
    }
  }
  __syncthreads();

  // ---- Phase D: O_head = P x V (K dim = 64 tokens, 2 steps of 32) ----
  // Wave owns M-strip mt and N-tiles n0..n0+7 (spanning 2 heads).
  {
    const int mt = wave >> 1;
    const int n0 = (wave & 1) * 8;
    v8f acc[8];
#pragma unroll
    for (int j = 0; j < 8; ++j) acc[j] = vzero;
#pragma unroll
    for (int hh = 0; hh < 2; ++hh) {
      int h = (n0 >> 2) + hh;            // n0=0 -> heads 0,1 ; n0=8 -> heads 2,3
#pragma unroll
      for (int kk = 0; kk < 2; ++kk) {
        v16bf pa = load_a(p_s + h * NTOK * PST, mt * 16, PST, kk * 32);  // hoisted
        v16bf vb[4];
#pragma unroll
        for (int jj = 0; jj < 4; ++jj)
          vb[jj] = load_b(vT, (h * 4 + jj) * 16, VST, kk * 32);  // B[n][c] = vT[c][n]
#pragma unroll
        for (int jj = 0; jj < 4; ++jj) {
          int nt = h * 4 + jj;
          acc[nt - n0] = wmma_bf16(pa, vb[jj], acc[nt - n0]);
        }
      }
    }
#pragma unroll
    for (int j = 0; j < 8; ++j) {
      int c = (n0 + j) * 16 + l;
#pragma unroll
      for (int r = 0; r < 8; ++r) {
        int t = mt * 16 + r + 8 * hi;
        o_s[t * AST + c] = (__bf16)acc[j][r];
      }
    }
  }
  __syncthreads();

  // ---- Phase E: Out = O x Wo^T + bo, scatter with window-reverse ----
  // Same 2 M-strip x 4 N-tile tiling + double-buffered weight stream.
  {
    const int m0 = (wave >> 2) * 2;
    const int n0 = (wave & 3) * 4;
    v8f acc0[4], acc1[4];
#pragma unroll
    for (int j = 0; j < 4; ++j) { acc0[j] = vzero; acc1[j] = vzero; }

    v16bf bfA[4], bfB[4];
#pragma unroll
    for (int j = 0; j < 4; ++j) bfA[j] = load_b(Wo, (n0 + j) * 16, DIM, 0);

#pragma unroll
    for (int kk = 0; kk < 8; kk += 2) {
      v16bf af0 = load_a(o_s, (m0 + 0) * 16, AST, kk * 32);
      v16bf af1 = load_a(o_s, (m0 + 1) * 16, AST, kk * 32);
#pragma unroll
      for (int j = 0; j < 4; ++j) bfB[j] = load_b(Wo, (n0 + j) * 16, DIM, (kk + 1) * 32);
      if (kk + 2 < 8)
        __builtin_prefetch(Wo + (size_t)(n0 * 16 + l) * DIM + (kk + 2) * 32, 0, 3);
#pragma unroll
      for (int j = 0; j < 4; ++j) {
        acc0[j] = wmma_bf16(af0, bfA[j], acc0[j]);
        acc1[j] = wmma_bf16(af1, bfA[j], acc1[j]);
      }
      af0 = load_a(o_s, (m0 + 0) * 16, AST, (kk + 1) * 32);
      af1 = load_a(o_s, (m0 + 1) * 16, AST, (kk + 1) * 32);
      if (kk + 2 < 8) {
#pragma unroll
        for (int j = 0; j < 4; ++j) bfA[j] = load_b(Wo, (n0 + j) * 16, DIM, (kk + 2) * 32);
      }
#pragma unroll
      for (int j = 0; j < 4; ++j) {
        acc0[j] = wmma_bf16(af0, bfB[j], acc0[j]);
        acc1[j] = wmma_bf16(af1, bfB[j], acc1[j]);
      }
    }

#pragma unroll
    for (int j = 0; j < 4; ++j) {
      int c = (n0 + j) * 16 + l;
      float bz = bo[c];
#pragma unroll
      for (int r = 0; r < 8; ++r) {
        int t0 = (m0 + 0) * 16 + r + 8 * hi;
        int t1 = (m0 + 1) * 16 + r + 8 * hi;
        int ya = y0 + (t0 >> 3), xa = x0 + (t0 & 7);
        int yb = y0 + (t1 >> 3), xb = x0 + (t1 & 7);
        out[img_base + ((size_t)c * IMG + ya) * IMG + xa] = acc0[j][r] + bz;
        out[img_base + ((size_t)c * IMG + yb) * IMG + xb] = acc1[j][r] + bz;
      }
    }
  }
}

// One-shot f32 -> bf16 conversion of the four 256x256 weight matrices into d_ws.
extern "C" __global__ void __launch_bounds__(256)
convert_weights(const float* __restrict__ wq, const float* __restrict__ wk,
                const float* __restrict__ wv, const float* __restrict__ wo,
                __bf16* __restrict__ dst) {
  int i = blockIdx.x * 256 + threadIdx.x;   // 0 .. 4*65536-1
  int m = i >> 16;
  int off = i & 65535;
  const float* s = (m == 0) ? wq : (m == 1) ? wk : (m == 2) ? wv : wo;
  dst[i] = (__bf16)s[off];
}

extern "C" void kernel_launch(void* const* d_in, const int* in_sizes, int n_in,
                              void* d_out, int out_size, void* d_ws, size_t ws_size,
                              hipStream_t stream) {
  const float* fa = (const float*)d_in[0];
  const float* fb = (const float*)d_in[1];
  const float* Wq = (const float*)d_in[2];
  const float* bq = (const float*)d_in[3];
  const float* Wk = (const float*)d_in[4];
  const float* bk = (const float*)d_in[5];
  const float* Wv = (const float*)d_in[6];
  const float* bv = (const float*)d_in[7];
  const float* Wo = (const float*)d_in[8];
  const float* bo = (const float*)d_in[9];

  float* out  = (float*)d_out;
  float* attn = out + (size_t)8 * DIM * IMG * IMG;   // second tuple element

  __bf16* wbf = (__bf16*)d_ws;                       // 4 * 65536 bf16 = 512 KB
  convert_weights<<<1024, 256, 0, stream>>>(Wq, Wk, Wv, Wo, wbf);

  const size_t smem_elems = (size_t)5 * NTOK * AST   // a, b, q, k, o
                          + (size_t)DIM * VST        // v^T
                          + (size_t)HEADS * NTOK * PST;
  const size_t smem = smem_elems * sizeof(__bf16);   // 242688 B < 320 KB/WGP

  win_attn_kernel<<<NWIN, 256, smem, stream>>>(
      fa, fb,
      wbf + 0,      bq,
      wbf + 65536,  bk,
      wbf + 131072, bv,
      wbf + 196608, bo,
      out, attn);
}
